// FullComboODETransition_88347477279462
// MI455X (gfx1250) — matrix operand order repaired and checked by reading, not compiled, not run on hardware
//
#include <hip/hip_runtime.h>
#include <hip/hip_bf16.h>

typedef __attribute__((ext_vector_type(16))) _Float16 v16h;
typedef __attribute__((ext_vector_type(8)))  _Float16 v8h;
typedef __attribute__((ext_vector_type(4)))  _Float16 v4h;
typedef __attribute__((ext_vector_type(8)))  float    v8f;
typedef __attribute__((ext_vector_type(4)))  float    v4f;

// Problem constants
constexpr int BATCH   = 16384;
constexpr int LATENT  = 128;
constexpr int U_DIM   = 30;
constexpr int NUM_PROD= 20;
constexpr int NUM_INJ = 10;
constexpr int FEAT_C  = 128;
constexpr int HID     = 64;
constexpr int ZU      = LATENT + U_DIM;      // 158
constexpr int ZUP     = 160;                 // padded to mult of 32
constexpr int WINP    = 288;                 // padded 286 -> 160(zu+pad) + 128(feat)
constexpr int YW      = NUM_PROD*2 + NUM_INJ;// 50

// Workspace layout (bytes)
constexpr size_t WS_A   = 0;                           // f32 [128][128]
constexpr size_t WS_WZ  = 65536;                       // f16 5*8*512
constexpr size_t WS_WI  = WS_WZ + 5*8*512*2;           // f16 5*1*512
constexpr size_t WS_W1  = WS_WI + 5*512*2;             // f16 20*9*4*512
constexpr size_t WS_W2  = WS_W1 + (size_t)20*9*4*512*2;// f16 20*2*512

// ---------------- prep: A = -(L L^T) ----------------
__global__ void compute_A_kernel(const float* __restrict__ L, float* __restrict__ A) {
    int idx = blockIdx.x * 256 + threadIdx.x;    // 16384 elements
    int i = idx >> 7, j = idx & 127;
    const float* li = L + i * 128;
    const float* lj = L + j * 128;
    float s = 0.f;
    #pragma unroll 8
    for (int k = 0; k < 128; ++k) s += li[k] * lj[k];
    A[idx] = -s;
}

// B-operand block layout: block of 512 halfs per (ktile,ntile);
// element e: lane = e>>4, half = e&15; n = lane&15; k = ((lane>>4)<<4) + half
__device__ __forceinline__ void decode_bslot(int e, int& k, int& n) {
    int lane = e >> 4, half = e & 15;
    n = lane & 15;
    k = ((lane >> 4) << 4) + half;
}

// Wz: [160][128]: rows 0..127 = A[n][k] (A symmetric), rows 128..157 = Bm[n][row-128]
__global__ void swizzle_wz_kernel(const float* __restrict__ A, const float* __restrict__ Bm,
                                  _Float16* __restrict__ dst) {
    int gid = blockIdx.x * 256 + threadIdx.x;    // 5*8*512
    int blk = gid >> 9, e = gid & 511;
    int ktile = blk >> 3, ntile = blk & 7;
    int k, n; decode_bslot(e, k, n);
    int K = ktile * 32 + k, N = ntile * 16 + n;
    float v = 0.f;
    if (K < LATENT)      v = A[N * 128 + K];
    else if (K < ZU)     v = Bm[N * U_DIM + (K - LATENT)];
    dst[gid] = (_Float16)v;
}

// Wi: [158][10] row-major, padded to [160][16]
__global__ void swizzle_wi_kernel(const float* __restrict__ Wi, _Float16* __restrict__ dst) {
    int gid = blockIdx.x * 256 + threadIdx.x;    // 5*512
    int ktile = gid >> 9, e = gid & 511;
    int k, n; decode_bslot(e, k, n);
    int K = ktile * 32 + k;
    float v = (K < ZU && n < NUM_INJ) ? Wi[K * NUM_INJ + n] : 0.f;
    dst[gid] = (_Float16)v;
}

// W1: [20][286][64]; padded K: 0..157 -> row K, 158..159 -> 0, 160..287 -> row K-2
__global__ void swizzle_w1_kernel(const float* __restrict__ W1, _Float16* __restrict__ dst) {
    int gid = blockIdx.x * 256 + threadIdx.x;    // 20*9*4*512
    int blk = gid >> 9, e = gid & 511;
    int p = blk / 36, rem = blk % 36;
    int ktile = rem >> 2, ntile = rem & 3;
    int k, n; decode_bslot(e, k, n);
    int K = ktile * 32 + k, N = ntile * 16 + n;
    float v = 0.f;
    int row = -1;
    if (K < ZU) row = K;
    else if (K >= ZUP) row = K - 2;              // 160..287 -> 158..285
    if (row >= 0) v = W1[((size_t)p * 286 + row) * HID + N];
    dst[gid] = (_Float16)v;
}

// W2: [20][64][2], padded N to 16
__global__ void swizzle_w2_kernel(const float* __restrict__ W2, _Float16* __restrict__ dst) {
    int gid = blockIdx.x * 256 + threadIdx.x;    // 20*2*512
    int blk = gid >> 9, e = gid & 511;
    int p = blk >> 1, ktile = blk & 1;
    int k, n; decode_bslot(e, k, n);
    int K = ktile * 32 + k;
    float v = (n < 2) ? W2[((size_t)p * HID + K) * 2 + n] : 0.f;
    dst[gid] = (_Float16)v;
}

// ---------------- fused forward ----------------
constexpr int WAVES = 4;
constexpr int ZU_STRIDE = 296;                   // halfs, bank-staggered row pitch
constexpr int H_STRIDE  = 72;
constexpr int LDS_ZU = 16 * ZU_STRIDE;           // 4736 halfs
constexpr int LDS_H  = 16 * H_STRIDE;            // 1152 halfs
constexpr int LDS_WAVE = LDS_ZU + LDS_H;         // 5888 halfs = 11776 B per wave

__device__ __forceinline__ v8f wmma16(v16h a, v16h b, v8f c) {
    return __builtin_amdgcn_wmma_f32_16x16x32_f16(false, a, false, b, (short)0, c, false, false);
}

// A-operand (16x32 f16), contiguous-k order matching B: lanes 0-15 M=lane hold K {0..15},
// lanes 16-31 M=lane-16 hold K {16..31} -> two 16B contiguous LDS chunks per lane.
__device__ __forceinline__ v16h load_a(const _Float16* rowBase, int rowStride, int ktile, int lane) {
    int m = lane & 15;
    int grp = lane >> 4;
    const _Float16* p = rowBase + m * rowStride + ktile * 32 + grp * 16;
    v8h lo = *(const v8h*)(p);
    v8h hi = *(const v8h*)(p + 8);
    v16h a;
    #pragma unroll
    for (int i = 0; i < 8; ++i) { a[i] = lo[i]; a[8 + i] = hi[i]; }
    return a;
}

__device__ __forceinline__ v16h load_b(const _Float16* buf, int blk, int lane) {
    return *(const v16h*)(buf + (size_t)blk * 512 + lane * 16);
}

__global__ void __launch_bounds__(WAVES * 32)
fused_forward_kernel(const float* __restrict__ zt, const float* __restrict__ ut,
                     const float* __restrict__ wf,
                     const float* __restrict__ b1, const float* __restrict__ b2,
                     const float* __restrict__ bi,
                     const _Float16* __restrict__ WzB, const _Float16* __restrict__ WiB,
                     const _Float16* __restrict__ W1B, const _Float16* __restrict__ W2B,
                     float* __restrict__ out) {
    const int lane = threadIdx.x & 31;
    const int wave = threadIdx.x >> 5;
    const int m0 = (blockIdx.x * WAVES + wave) * 16;

    __shared__ _Float16 lds[WAVES * LDS_WAVE];
    _Float16* zuL = lds + wave * LDS_WAVE;
    _Float16* hL  = zuL + LDS_ZU;

    // ---- stage [zt | ut | 0pad] rows m0..m0+15 cols 0..159 as f16 ----
    {
        int row = lane >> 1;
        int c0 = (lane & 1) * 80;
        _Float16* dst = zuL + row * ZU_STRIDE + c0;
        const float* ztRow = zt + (size_t)(m0 + row) * LATENT;
        const float* utRow = ut + (size_t)(m0 + row) * U_DIM;
        #pragma unroll 4
        for (int i = 0; i < 80; ++i) {
            int c = c0 + i;
            float v = (c < LATENT) ? ztRow[c] : ((c < ZU) ? utRow[c - LATENT] : 0.f);
            dst[i] = (_Float16)v;
        }
    }

    const int n  = lane & 15;
    const int mb = (lane >> 4) * 8;              // C/D layout: m = r + mb
    v8f vzero = {};

    // ---- zt1 = zt + [zt|ut] @ Wz  (8 ntiles x 5 ktiles) ----
    v8f accZ[8];
    #pragma unroll
    for (int t = 0; t < 8; ++t) accZ[t] = vzero;
    #pragma unroll
    for (int k = 0; k < 5; ++k) {
        v16h a = load_a(zuL, ZU_STRIDE, k, lane);
        #pragma unroll
        for (int t = 0; t < 8; ++t) {
            v16h b = load_b(WzB, k * 8 + t, lane);
            accZ[t] = wmma16(a, b, accZ[t]);
        }
    }
    // epilogue: add zt (f32), write f32 zt1 to out, write f16 zt1 back to LDS zu
    #pragma unroll
    for (int t = 0; t < 8; ++t) {
        #pragma unroll
        for (int r = 0; r < 8; ++r) {
            int m = mb + r;
            int col = t * 16 + n;
            size_t gi = (size_t)(m0 + m) * LATENT + col;
            float v = accZ[t][r] + zt[gi];
            out[gi] = v;
            zuL[m * ZU_STRIDE + col] = (_Float16)v;
        }
    }

    // cache the 5 zu A-operands (shared by injector + all wells' K<160)
    v16h aZu[5];
    #pragma unroll
    for (int k = 0; k < 5; ++k) aZu[k] = load_a(zuL, ZU_STRIDE, k, lane);

    float* yt = out + (size_t)BATCH * LATENT;

    // ---- injector head: [16,160] x [160,16pad] ----
    {
        v8f acc = vzero;
        #pragma unroll
        for (int k = 0; k < 5; ++k) acc = wmma16(aZu[k], load_b(WiB, k, lane), acc);
        if (n < NUM_INJ) {
            float bv = bi[n];
            #pragma unroll
            for (int r = 0; r < 8; ++r) {
                int m = mb + r;
                yt[(size_t)(m0 + m) * YW + NUM_PROD * 2 + n] = acc[r] + bv;
            }
        }
    }

    // ---- per-well MLP heads ----
    for (int p = 0; p < NUM_PROD; ++p) {
        // stage well_features[:, p, :] into cols 160..287 (coalesced f32x4 -> f16x4)
        {
            int row = lane >> 1;
            int c0 = (lane & 1) * 64;
            const float* src = wf + (((size_t)(m0 + row) * NUM_PROD + p) * FEAT_C) + c0;
            _Float16* dst = zuL + row * ZU_STRIDE + ZUP + c0;
            #pragma unroll
            for (int i = 0; i < 16; ++i) {
                v4f v = *(const v4f*)(src + i * 4);
                v4h h4;
                h4[0] = (_Float16)v[0]; h4[1] = (_Float16)v[1];
                h4[2] = (_Float16)v[2]; h4[3] = (_Float16)v[3];
                *(v4h*)(dst + i * 4) = h4;
            }
        }
        // layer1: [16,288] x [288,64] -> 9 ktiles x 4 ntiles
        v8f acc1[4];
        #pragma unroll
        for (int t = 0; t < 4; ++t) acc1[t] = vzero;
        #pragma unroll
        for (int k = 0; k < 9; ++k) {
            v16h a = (k < 5) ? aZu[k] : load_a(zuL, ZU_STRIDE, k, lane);
            #pragma unroll
            for (int t = 0; t < 4; ++t) {
                v16h b = load_b(W1B, (p * 9 + k) * 4 + t, lane);
                acc1[t] = wmma16(a, b, acc1[t]);
            }
        }
        // bias + relu, store h as f16 to LDS
        #pragma unroll
        for (int t = 0; t < 4; ++t) {
            float bv = b1[p * HID + t * 16 + n];
            #pragma unroll
            for (int r = 0; r < 8; ++r) {
                float v = acc1[t][r] + bv;
                v = v > 0.f ? v : 0.f;
                hL[(mb + r) * H_STRIDE + t * 16 + n] = (_Float16)v;
            }
        }
        // layer2: [16,64] x [64,16pad] -> 2 ktiles
        v8f acc2 = vzero;
        #pragma unroll
        for (int k = 0; k < 2; ++k) {
            v16h a = load_a(hL, H_STRIDE, k, lane);
            acc2 = wmma16(a, load_b(W2B, p * 2 + k, lane), acc2);
        }
        if (n < 2) {
            float bv = b2[p * 2 + n];
            #pragma unroll
            for (int r = 0; r < 8; ++r) {
                int m = mb + r;
                yt[(size_t)(m0 + m) * YW + p * 2 + n] = acc2[r] + bv;
            }
        }
    }
}

extern "C" void kernel_launch(void* const* d_in, const int* in_sizes, int n_in,
                              void* d_out, int out_size, void* d_ws, size_t ws_size,
                              hipStream_t stream) {
    (void)in_sizes; (void)n_in; (void)out_size; (void)ws_size;
    const float* zt = (const float*)d_in[0];
    // d_in[1] = dt (int, unused: euler step h=1 is baked into the math)
    const float* ut = (const float*)d_in[2];
    const float* wf = (const float*)d_in[3];
    const float* L  = (const float*)d_in[4];
    const float* Bm = (const float*)d_in[5];
    const float* W1 = (const float*)d_in[6];
    const float* b1 = (const float*)d_in[7];
    const float* W2 = (const float*)d_in[8];
    const float* b2 = (const float*)d_in[9];
    const float* Wi = (const float*)d_in[10];
    const float* bi = (const float*)d_in[11];

    char* ws = (char*)d_ws;
    float*     A    = (float*)(ws + WS_A);
    _Float16*  WzB  = (_Float16*)(ws + WS_WZ);
    _Float16*  WiB  = (_Float16*)(ws + WS_WI);
    _Float16*  W1B  = (_Float16*)(ws + WS_W1);
    _Float16*  W2B  = (_Float16*)(ws + WS_W2);
    float* out = (float*)d_out;

    compute_A_kernel<<<64, 256, 0, stream>>>(L, A);
    swizzle_wz_kernel<<<80, 256, 0, stream>>>(A, Bm, WzB);
    swizzle_wi_kernel<<<10, 256, 0, stream>>>(Wi, WiB);
    swizzle_w1_kernel<<<1440, 256, 0, stream>>>(W1, W1B);
    swizzle_w2_kernel<<<80, 256, 0, stream>>>(W2, W2B);

    fused_forward_kernel<<<BATCH / (WAVES * 16), WAVES * 32, 0, stream>>>(
        zt, ut, wf, b1, b2, bi, WzB, WiB, W1B, W2B, out);
}
